// TakingHeadAttention_30588757082372
// MI455X (gfx1250) — matrix operand order repaired and checked by reading, not compiled, not run on hardware
//
#include <hip/hip_runtime.h>
#include <hip/hip_bf16.h>

#define D   1024
#define H   16
#define HD  64
#define BB  2
#define LL  2048
#define NTOK (BB*LL)

typedef __bf16 v8bf  __attribute__((ext_vector_type(8)));
typedef __bf16 v16bf __attribute__((ext_vector_type(16)));
typedef float  v8f   __attribute__((ext_vector_type(8)));
typedef unsigned int uint4v __attribute__((ext_vector_type(4)));
typedef int          int4v  __attribute__((ext_vector_type(4)));
typedef int          int8v  __attribute__((ext_vector_type(8)));

__device__ __forceinline__ unsigned short f2bf(float f) {
  unsigned int u = __builtin_bit_cast(unsigned int, f);
  u += 0x7fffu + ((u >> 16) & 1u);          // round-to-nearest-even
  return (unsigned short)(u >> 16);
}

// Build a 16-element bf16 fragment from two contiguous 16-byte halves.
__device__ __forceinline__ v16bf load_frag(const unsigned short* p, int lo, int hi) {
  v8bf a = *(const v8bf*)(p + lo);
  v8bf b = *(const v8bf*)(p + hi);
  return __builtin_shufflevector(a, b, 0,1,2,3,4,5,6,7,8,9,10,11,12,13,14,15);
}

__device__ __forceinline__ v8f wmma_bf16(v16bf a, v16bf b, v8f c) {
  return __builtin_amdgcn_wmma_f32_16x16x32_bf16(false, a, false, b, (short)0, c, false, false);
}

// CDNA5 async global->LDS copy, 16B per lane, tracked by ASYNCcnt.
__device__ __forceinline__ void async_copy16(unsigned lds_off, const void* gaddr) {
  asm volatile("global_load_async_to_lds_b128 %0, %1, off"
               :: "v"(lds_off), "v"((unsigned long long)gaddr) : "memory");
}
__device__ __forceinline__ void wait_async0() {
  asm volatile("s_wait_asynccnt 0x0" ::: "memory");
}
__device__ __forceinline__ unsigned lds_off_of(const void* p) {
  // flat LDS-aperture address: low 32 bits are the workgroup-relative LDS offset
  return (unsigned)(unsigned long long)p;
}

// ---------------------------------------------------------------------------
// Tensor Data Mover: one-shot 2-D tile DMA global -> LDS (TENSORcnt-tracked).
// D# per CDNA5 ISA: group0 {count=1, lds_addr,
// global_addr[56:0], type=2}; group1 {data_size=2B, tensor dims, tile dims,
// dim0 stride}; groups 2/3 zero (2-D tile).
// tile_k: elements along contiguous dim; tile_rows: rows; row_stride: elements.
// ---------------------------------------------------------------------------
__device__ __forceinline__ void tdm_load_tile_2d(unsigned lds_addr, const void* gaddr,
                                                 unsigned tile_k, unsigned tile_rows,
                                                 unsigned tensor_k, unsigned tensor_rows,
                                                 unsigned row_stride) {
  const unsigned long long ga = (unsigned long long)gaddr;
  uint4v g0;
  g0[0] = 1u;                                                   // count=1, user mode
  g0[1] = lds_addr;                                             // lds_addr [63:32]
  g0[2] = (unsigned)(ga & 0xffffffffu);                         // global_addr [95:64]
  g0[3] = (unsigned)((ga >> 32) & 0x01ffffffu) | (2u << 30);    // addr[56:32] | type=2
  int8v g1;
  g1[0] = (int)(1u << 16);                                      // data_size=1 -> 2 bytes
  g1[1] = (int)((tensor_k & 0xffffu) << 16);                    // tensor_dim0[15:0]
  g1[2] = (int)(((tensor_k >> 16) & 0xffffu) |
                ((tensor_rows & 0xffffu) << 16));               // dim0 hi | dim1 lo
  g1[3] = (int)(((tensor_rows >> 16) & 0xffffu) |
                ((tile_k & 0xffffu) << 16));                    // dim1 hi | tile_dim0
  g1[4] = (int)(tile_rows & 0xffffu);                           // tile_dim1 (tile_dim2=0)
  g1[5] = (int)row_stride;                                      // tensor_dim0_stride lo32
  g1[6] = 0;                                                    // stride hi | dim1_stride lo
  g1[7] = 0;
  const int4v z4 = {0, 0, 0, 0};
#if defined(__clang_major__) && (__clang_major__ >= 23)
  const int8v z8 = {0, 0, 0, 0, 0, 0, 0, 0};
  __builtin_amdgcn_tensor_load_to_lds(g0, g1, z4, z4, z8, 0);
#else
  __builtin_amdgcn_tensor_load_to_lds(g0, g1, z4, z4, 0);
#endif
}

// ---------------------------------------------------------------------------
// One-shot fp32 -> bf16 conversion (hoists all cvt VALU out of GEMM hot loops)
// ---------------------------------------------------------------------------
__global__ void __launch_bounds__(256)
cvt_bf16_kernel(const float* __restrict__ src, unsigned short* __restrict__ dst, int n4) {
  const int i = blockIdx.x * 256 + threadIdx.x;
  if (i < n4) {
    const float4 f = ((const float4*)src)[i];
    ushort4 o;
    o.x = f2bf(f.x); o.y = f2bf(f.y); o.z = f2bf(f.z); o.w = f2bf(f.w);
    ((ushort4*)dst)[i] = o;
  }
}

// ---------------------------------------------------------------------------
// C[n,o] = sum_k A[n,k] * W[o,k] + bias[o], all-bf16 operands.
// Block tile 64(M) x 128(N), K-step 64, double-buffered staging:
//   A tile  (64x64)  : per-lane global_load_async_to_lds_b128 (ASYNCcnt)
//   W tile (128x64)  : one TDM tensor_load_to_lds per buffer  (TENSORcnt)
// 8 waves: 4(M) x 2(N), each computing 16x64.
// OMODE 0: fp32 out [NTOK,D]
// OMODE 1: bf16 out [B,H,L,HD]          -> q/k head layout
// OMODE 2: bf16 out [B,H,HD,L] (transp) -> v for A*V B-operand
// ---------------------------------------------------------------------------
template<int OMODE>
__global__ void __launch_bounds__(256)
gemm_bf16_kernel(const unsigned short* __restrict__ A, const unsigned short* __restrict__ W,
                 const float* __restrict__ bias, void* __restrict__ out)
{
  __shared__ unsigned short Abf[2][64 * 64];    // 16 KB
  __shared__ unsigned short Bbf[2][128 * 64];   // 32 KB

  const int tid    = threadIdx.x;
  const int wave   = tid >> 5;
  const int ln     = tid & 31;
  const int hlf    = ln >> 4;
  const int lane15 = ln & 15;
  const int mw     = (wave >> 1) * 16;
  const int nw     = (wave & 1) * 64;
  const int rowBase = blockIdx.y * 64;
  const int oBase   = blockIdx.x * 128;

  v8f acc[4];
  for (int nt = 0; nt < 4; ++nt) acc[nt] = (v8f){0.f,0.f,0.f,0.f,0.f,0.f,0.f,0.f};

  auto issue = [&](int bi, int k0) {
    // A tile 64x64 bf16 = 512 x 16B chunks -> 2 per thread (async to LDS)
    for (int i = 0; i < 2; ++i) {
      const int e = tid + i * 256;
      const int r = e >> 3, c = (e & 7) * 8;
      async_copy16(lds_off_of(&Abf[bi][r * 64 + c]),
                   A + (size_t)(rowBase + r) * D + k0 + c);
    }
    // W tile 128x64 bf16: single TDM descriptor, issued by wave 0 only
    if (wave == 0)
      tdm_load_tile_2d(lds_off_of(&Bbf[bi][0]),
                       W + (size_t)oBase * D + k0,
                       /*tile_k=*/64, /*tile_rows=*/128,
                       /*tensor_k=*/D, /*tensor_rows=*/128,
                       /*row_stride=*/D);
  };

  issue(0, 0);
  for (int it = 0; it < D / 64; ++it) {
    const int bi = it & 1;
    wait_async0();                          // this wave's A-tile copies landed
    __builtin_amdgcn_s_wait_tensorcnt(0);   // wave0's W-tile DMA landed
    __syncthreads();                        // everyone sees both tiles
    if (it + 1 < D / 64) issue(bi ^ 1, (it + 1) * 64);   // overlap DMA w/ WMMA

    for (int kk = 0; kk < 2; ++kk) {
      const v16bf a = load_frag(&Abf[bi][(mw + lane15) * 64 + kk * 32],
                                hlf * 8, 16 + hlf * 8);
      for (int nt = 0; nt < 4; ++nt) {
        const v16bf b = load_frag(&Bbf[bi][(nw + nt * 16 + lane15) * 64 + kk * 32],
                                  hlf * 8, 16 + hlf * 8);
        acc[nt] = wmma_bf16(a, b, acc[nt]);
      }
    }
  }

  for (int nt = 0; nt < 4; ++nt) {
    const int o  = oBase + nw + nt * 16 + lane15;
    const float bsv = bias[o];
    for (int r = 0; r < 8; ++r) {
      const int row = rowBase + mw + r + 8 * hlf;   // C frag: VGPR r -> M=r / r+8
      const float val = acc[nt][r] + bsv;
      if constexpr (OMODE == 0) {
        ((float*)out)[(size_t)row * D + o] = val;
      } else {
        const int b_ = row >> 11, l = row & (LL - 1);
        const int h = o >> 6, hd = o & 63;
        if constexpr (OMODE == 1)
          ((unsigned short*)out)[((size_t)(b_ * H + h) * LL + l) * HD + hd] = f2bf(val);
        else
          ((unsigned short*)out)[((size_t)(b_ * H + h) * HD + hd) * LL + l] = f2bf(val);
      }
    }
  }
}

// ---------------------------------------------------------------------------
// Fused talking-heads flash attention.
// grid = (L/16, B); block = 256 (8 waves); wave w owns heads {2w, 2w+1}.
// qh,kh: bf16 [B,H,L,HD]; vT: bf16 [B,H,HD,L]; pre: bf16 [B,L,D] output.
// ---------------------------------------------------------------------------
__global__ void __launch_bounds__(256)
attn_kernel(const unsigned short* __restrict__ qh,
            const unsigned short* __restrict__ kh,
            const unsigned short* __restrict__ vT,
            const float* __restrict__ P,
            unsigned short* __restrict__ pre)
{
  __shared__ float S_lds[H * 16 * 32];   // [h][l][m]  32 KB
  __shared__ float P_lds[H * H];

  const int tid    = threadIdx.x;
  const int wave   = tid >> 5;
  const int ln     = tid & 31;
  const int hlf    = ln >> 4;
  const int lane15 = ln & 15;
  const int b  = blockIdx.y;
  const int l0 = blockIdx.x * 16;
  const int g0 = wave * 2;

  P_lds[tid] = P[tid];                   // 256 == H*H
  __syncthreads();

  // Persistent Q fragments: 2 heads x 2 K-steps (d 0-31, 32-63)
  v16bf qf[2][2];
  for (int hh = 0; hh < 2; ++hh) {
    const unsigned short* qp =
        qh + ((size_t)((b * H + g0 + hh) * LL) + l0 + lane15) * HD;
    qf[hh][0] = load_frag(qp,      hlf * 8, 16 + hlf * 8);
    qf[hh][1] = load_frag(qp, 32 + hlf * 8, 48 + hlf * 8);
  }

  v8f oacc[2][4];
  for (int hh = 0; hh < 2; ++hh)
    for (int dt = 0; dt < 4; ++dt)
      oacc[hh][dt] = (v8f){0.f,0.f,0.f,0.f,0.f,0.f,0.f,0.f};
  float m_run[2] = {-3.0e38f, -3.0e38f};
  float s_run[2] = {0.f, 0.f};

  for (int m0 = 0; m0 < LL; m0 += 32) {
    if (m0 + 32 < LL) {   // prefetch next chunk of K and V into near caches
      __builtin_prefetch(kh + ((size_t)((b * H + g0) * LL) + m0 + 32 + lane15) * HD, 0, 3);
      __builtin_prefetch(vT + ((size_t)((b * H + g0) * HD) + lane15) * LL + m0 + 32, 0, 3);
    }
    // ---- raw per-head logits S = (Q K^T) / sqrt(HD) into LDS ----
    for (int hh = 0; hh < 2; ++hh) {
      const int h = g0 + hh;
      for (int mt = 0; mt < 2; ++mt) {
        const unsigned short* kp =
            kh + ((size_t)((b * H + h) * LL) + m0 + mt * 16 + lane15) * HD;
        const v16bf b0 = load_frag(kp,      hlf * 8, 16 + hlf * 8);
        const v16bf b1 = load_frag(kp, 32 + hlf * 8, 48 + hlf * 8);
        v8f s = (v8f){0.f,0.f,0.f,0.f,0.f,0.f,0.f,0.f};
        s = wmma_bf16(qf[hh][0], b0, s);
        s = wmma_bf16(qf[hh][1], b1, s);
        for (int r = 0; r < 8; ++r)
          S_lds[h * 512 + (r + 8 * hlf) * 32 + mt * 16 + lane15] = s[r] * 0.125f;
      }
    }
    __syncthreads();

    // ---- talking-heads mix: mixed[g] = sum_h P[h,g] * S[h] ----
    // lane covers (l = lane15, m in A-fragment order): j<8 -> m=hlf*8+j,
    // j>=8 -> m=16+hlf*8+(j-8). Vectorized as float4 LDS loads.
    float mix0[16], mix1[16];
    for (int j = 0; j < 16; ++j) { mix0[j] = 0.f; mix1[j] = 0.f; }
    const int base0 = lane15 * 32 + hlf * 8;
    const int base1 = lane15 * 32 + 16 + hlf * 8;
    for (int h = 0; h < H; ++h) {
      const float p0 = P_lds[h * H + g0];
      const float p1 = P_lds[h * H + g0 + 1];
      const float4 x0 = *(const float4*)&S_lds[h * 512 + base0];
      const float4 x1 = *(const float4*)&S_lds[h * 512 + base0 + 4];
      const float4 x2 = *(const float4*)&S_lds[h * 512 + base1];
      const float4 x3 = *(const float4*)&S_lds[h * 512 + base1 + 4];
      const float xs[8] = {x0.x,x0.y,x0.z,x0.w, x1.x,x1.y,x1.z,x1.w};
      const float ys[8] = {x2.x,x2.y,x2.z,x2.w, x3.x,x3.y,x3.z,x3.w};
#pragma unroll
      for (int j = 0; j < 8; ++j) {
        mix0[j]     = fmaf(p0, xs[j], mix0[j]);
        mix1[j]     = fmaf(p1, xs[j], mix1[j]);
        mix0[j + 8] = fmaf(p0, ys[j], mix0[j + 8]);
        mix1[j + 8] = fmaf(p1, ys[j], mix1[j + 8]);
      }
    }
    __syncthreads();   // S_lds free for next chunk

    // ---- online softmax + A*V accumulate ----
    for (int hh = 0; hh < 2; ++hh) {
      const float* mixp = hh ? mix1 : mix0;
      float cm = -3.0e38f;
      for (int j = 0; j < 16; ++j) cm = fmaxf(cm, mixp[j]);
      cm = fmaxf(cm, __shfl_xor(cm, 16, 32));          // row spans lane pair ln/ln^16
      const float nm   = fmaxf(m_run[hh], cm);
      const float corr = __expf(m_run[hh] - nm);
      m_run[hh] = nm;

      union { v16bf v; unsigned short u[16]; } af;     // attn A-fragment, in-lane
      float csum = 0.f;
      for (int j = 0; j < 16; ++j) {
        const float p = __expf(mixp[j] - nm);
        csum += p;
        af.u[j] = f2bf(p);
      }
      csum += __shfl_xor(csum, 16, 32);
      s_run[hh] = s_run[hh] * corr + csum;

      float cf[8];
      for (int r = 0; r < 8; ++r) cf[r] = __shfl(corr, r + 8 * hlf, 32);
      for (int dt = 0; dt < 4; ++dt) {
        for (int r = 0; r < 8; ++r) oacc[hh][dt][r] *= cf[r];
        const unsigned short* vp =
            vT + ((size_t)((b * H + g0 + hh) * HD) + dt * 16 + lane15) * LL + m0;
        const v16bf bv = load_frag(vp, hlf * 8, 16 + hlf * 8);
        oacc[hh][dt] = wmma_bf16(af.v, bv, oacc[hh][dt]);
      }
    }
  }

  // ---- normalize and store pre-projection output (bf16 [B,L,D]) ----
  for (int hh = 0; hh < 2; ++hh) {
    float sf[8];
    for (int r = 0; r < 8; ++r)
      sf[r] = 1.0f / __shfl(s_run[hh], r + 8 * hlf, 32);
    for (int dt = 0; dt < 4; ++dt)
      for (int r = 0; r < 8; ++r) {
        const int row = l0 + r + 8 * hlf;
        const float val = oacc[hh][dt][r] * sf[r];
        pre[((size_t)(b * LL + row)) * D + (g0 + hh) * HD + dt * 16 + lane15] = f2bf(val);
      }
  }
}

// ---------------------------------------------------------------------------
extern "C" void kernel_launch(void* const* d_in, const int* in_sizes, int n_in,
                              void* d_out, int out_size, void* d_ws, size_t ws_size,
                              hipStream_t stream) {
  const float* q  = (const float*)d_in[0];
  const float* k  = (const float*)d_in[1];
  const float* v  = (const float*)d_in[2];
  const float* Wq = (const float*)d_in[3];
  const float* bq = (const float*)d_in[4];
  const float* Wk = (const float*)d_in[5];
  const float* bk = (const float*)d_in[6];
  const float* Wv = (const float*)d_in[7];
  const float* bv = (const float*)d_in[8];
  const float* Wo = (const float*)d_in[9];
  const float* bo = (const float*)d_in[10];
  const float* P  = (const float*)d_in[11];

  const size_t nQKV = (size_t)NTOK * D;     // 4M elements
  const size_t nW   = (size_t)D * D;        // 1M elements

  unsigned short* p   = (unsigned short*)d_ws;
  unsigned short* qbf = p;  p += nQKV;
  unsigned short* kbf = p;  p += nQKV;
  unsigned short* vbf = p;  p += nQKV;
  unsigned short* wqb = p;  p += nW;
  unsigned short* wkb = p;  p += nW;
  unsigned short* wvb = p;  p += nW;
  unsigned short* wob = p;  p += nW;
  unsigned short* qhh = p;  p += nQKV;      // [B,H,L,HD]
  unsigned short* khh = p;  p += nQKV;
  unsigned short* vT  = p;  p += nQKV;      // [B,H,HD,L]
  unsigned short* pre = p;  p += nQKV;      // [B,L,D]

  // one-shot fp32 -> bf16
  cvt_bf16_kernel<<<(int)(nQKV / 4 + 255) / 256, 256, 0, stream>>>(q,  qbf, (int)(nQKV / 4));
  cvt_bf16_kernel<<<(int)(nQKV / 4 + 255) / 256, 256, 0, stream>>>(k,  kbf, (int)(nQKV / 4));
  cvt_bf16_kernel<<<(int)(nQKV / 4 + 255) / 256, 256, 0, stream>>>(v,  vbf, (int)(nQKV / 4));
  cvt_bf16_kernel<<<(int)(nW / 4 + 255) / 256,   256, 0, stream>>>(Wq, wqb, (int)(nW / 4));
  cvt_bf16_kernel<<<(int)(nW / 4 + 255) / 256,   256, 0, stream>>>(Wk, wkb, (int)(nW / 4));
  cvt_bf16_kernel<<<(int)(nW / 4 + 255) / 256,   256, 0, stream>>>(Wv, wvb, (int)(nW / 4));
  cvt_bf16_kernel<<<(int)(nW / 4 + 255) / 256,   256, 0, stream>>>(Wo, wob, (int)(nW / 4));

  const dim3 ggrid(D / 128, NTOK / 64);     // (8, 64)
  gemm_bf16_kernel<1><<<ggrid, 256, 0, stream>>>(qbf, wqb, bq, qhh);
  gemm_bf16_kernel<1><<<ggrid, 256, 0, stream>>>(kbf, wkb, bk, khh);
  gemm_bf16_kernel<2><<<ggrid, 256, 0, stream>>>(vbf, wvb, bv, vT);

  attn_kernel<<<dim3(LL / 16, BB), 256, 0, stream>>>(qhh, khh, vT, P, pre);

  gemm_bf16_kernel<0><<<ggrid, 256, 0, stream>>>(pre, wob, bo, (float*)d_out);
}
